// MoEUpperNet_10797547782496
// MI455X (gfx1250) — compile-verified, matrix operand order checked by reading, not run on hardware
//
#include <hip/hip_runtime.h>
#include <hip/hip_bf16.h>
#include <math.h>

// Problem constants
#define BB     16
#define NLAY   12
#define L1     197
#define LFEAT  196
#define DDIM   768
#define EEXP   4
#define KTOP   3
#define CCLS   21
#define IMG    224

typedef __attribute__((ext_vector_type(16))) __bf16 v16bf;
typedef __attribute__((ext_vector_type(8)))  float  v8f;

// ---------------------------------------------------------------------------
// Kernel 1: gate scores.  grid = B*NL blocks of 64 threads.
// scoreBuf[(b*NL+nl)*E + e] = cls(b,nl) . gate_W[:,e] + gate_b[e]
// ---------------------------------------------------------------------------
__global__ void gate_kernel(const float* __restrict__ features,
                            const float* __restrict__ gate_W,
                            const float* __restrict__ gate_b,
                            float* __restrict__ scoreBuf) {
    int bn = blockIdx.x;                                   // b*NL + nl
    const float* cls = features + (size_t)bn * L1 * DDIM;  // row l=0 (CLS)
    __shared__ float red[64][EEXP];
    float acc[EEXP] = {0.f, 0.f, 0.f, 0.f};
    for (int d = threadIdx.x; d < DDIM; d += 64) {
        float f = cls[d];
#pragma unroll
        for (int e = 0; e < EEXP; ++e) acc[e] += f * gate_W[d * EEXP + e];
    }
#pragma unroll
    for (int e = 0; e < EEXP; ++e) red[threadIdx.x][e] = acc[e];
    __syncthreads();
    if (threadIdx.x < EEXP) {
        float s = gate_b[threadIdx.x];
        for (int t = 0; t < 64; ++t) s += red[t][threadIdx.x];
        scoreBuf[bn * EEXP + threadIdx.x] = s;
    }
}

// ---------------------------------------------------------------------------
// Kernel 2: per-(b,e) softmax over NL, top-3, softmax of the top values.
// One block of 64 threads; thread t handles (b,e) = (t/4, t%4).
// ---------------------------------------------------------------------------
__global__ void topk_kernel(const float* __restrict__ scoreBuf,
                            int* __restrict__ idxBuf,
                            float* __restrict__ wBuf) {
    int t = threadIdx.x;
    if (t >= BB * EEXP) return;
    int b = t >> 2, e = t & 3;
    float s[NLAY];
    float m = -1e30f;
    for (int nl = 0; nl < NLAY; ++nl) {
        s[nl] = scoreBuf[(b * NLAY + nl) * EEXP + e];
        m = fmaxf(m, s[nl]);
    }
    float sum = 0.f;
    for (int nl = 0; nl < NLAY; ++nl) { s[nl] = expf(s[nl] - m); sum += s[nl]; }
    float inv = 1.f / sum;
    for (int nl = 0; nl < NLAY; ++nl) s[nl] *= inv;   // gate_prob over NL

    int   id[KTOP];
    float tv[KTOP];
    bool used[NLAY];
    for (int nl = 0; nl < NLAY; ++nl) used[nl] = false;
    for (int k = 0; k < KTOP; ++k) {
        int best = -1; float bv = -1e30f;
        for (int nl = 0; nl < NLAY; ++nl)
            if (!used[nl] && s[nl] > bv) { bv = s[nl]; best = nl; }
        used[best] = true;
        id[k] = best; tv[k] = bv;
    }
    // softmax over the K top values (tv[0] is the max)
    float ws = 0.f, wv[KTOP];
    for (int k = 0; k < KTOP; ++k) { wv[k] = expf(tv[k] - tv[0]); ws += wv[k]; }
    float wi = 1.f / ws;
    for (int k = 0; k < KTOP; ++k) {
        idxBuf[t * KTOP + k] = id[k];
        wBuf[t * KTOP + k]   = wv[k] * wi;
    }
}

// ---------------------------------------------------------------------------
// Kernel 3: fused layer-mix + expert GEMM using WMMA bf16.
// grid = B*E blocks of 256 threads (8 waves).  Block (b,e):
//   mixed[l,d] = sum_k w_k * feats[b, idx_k, 1+l, d]      (built per-fragment)
//   logits[b,e,l,c] = mixed @ expert_W[e] + expert_b[e]   (196x768 x 768x21)
// B fragments (expert_W in bf16, WMMA layout) are pre-packed into LDS.
// ---------------------------------------------------------------------------
#define KSTEPS (DDIM / 32)     // 24
#define NTILES 2               // cols 0..15 and 16..31 (21 valid)
#define MTILES 13              // ceil(196/16)

__global__ void moe_gemm_kernel(const float* __restrict__ features,
                                const float* __restrict__ expert_W,
                                const float* __restrict__ expert_b,
                                const int*   __restrict__ idxBuf,
                                const float* __restrict__ wBuf,
                                float* __restrict__ logits) {
    const int be = blockIdx.x;          // b*E + e
    const int b  = be >> 2;
    const int e  = be & 3;

    // ---- Stage 1: pack expert_W[e] (768x21, row-major) into LDS as bf16
    // WMMA B fragments.  Layout: Bs[ks][nt][lane][j], lane holds column
    // N = nt*16 + (lane&15); j in 0..7 -> K = ks*32 + (lane>>4)*8 + j,
    // j in 8..15 -> K = ks*32 + 16 + (lane>>4)*8 + (j-8).
    __shared__ __bf16 Bs[KSTEPS * NTILES * 32 * 16];   // 48 KB
    for (int tix = threadIdx.x; tix < KSTEPS * NTILES * 32 * 16; tix += 256) {
        int j    = tix & 15;
        int lane = (tix >> 4) & 31;
        int nt   = (tix >> 9) & 1;
        int ks   = tix >> 10;
        int c    = nt * 16 + (lane & 15);
        int k    = ks * 32 + ((lane >> 4) * 8) + (j & 7) + ((j >> 3) * 16);
        float v  = (c < CCLS) ? expert_W[((size_t)e * DDIM + k) * CCLS + c] : 0.f;
        Bs[tix] = (__bf16)v;
    }
    __syncthreads();

    const int lane  = threadIdx.x & 31;
    const int wid   = threadIdx.x >> 5;
    const int row   = lane & 15;
    const int khalf = (lane >> 4) * 8;       // 0 or 8
    const int rbase = (lane >> 4) * 8;       // accumulator row base
    const int cbase = lane & 15;             // accumulator column

    const int   i0 = idxBuf[be * KTOP + 0];
    const int   i1 = idxBuf[be * KTOP + 1];
    const int   i2 = idxBuf[be * KTOP + 2];
    const float w0 = wBuf[be * KTOP + 0];
    const float w1 = wBuf[be * KTOP + 1];
    const float w2 = wBuf[be * KTOP + 2];

    // Loop-invariant bias per lane (column fixed by lane id)
    const int   c1    = 16 + cbase;
    const bool  c1ok  = (c1 < CCLS);
    const float bias0 = expert_b[e * CCLS + cbase];
    const float bias1 = c1ok ? expert_b[e * CCLS + c1] : 0.f;

    for (int mt = wid; mt < MTILES; mt += 8) {
        int l  = mt * 16 + row;
        int lc = l < LFEAT ? l : (LFEAT - 1);      // clamp; bad rows masked at store
        const float* p0 = features + (((size_t)b * NLAY + i0) * L1 + 1 + lc) * DDIM;
        const float* p1 = features + (((size_t)b * NLAY + i1) * L1 + 1 + lc) * DDIM;
        const float* p2 = features + (((size_t)b * NLAY + i2) * L1 + 1 + lc) * DDIM;

        v8f acc0 = {};
        v8f acc1 = {};
        for (int ks = 0; ks < KSTEPS; ++ks) {
            int kb = ks * 32 + khalf;
            // stream-ahead prefetch of the three feature rows (speculative)
            __builtin_prefetch(p0 + kb + 64, 0, 0);
            __builtin_prefetch(p1 + kb + 64, 0, 0);
            __builtin_prefetch(p2 + kb + 64, 0, 0);
            v16bf a;
#pragma unroll
            for (int j = 0; j < 8; ++j) {
                float f = w0 * p0[kb + j] + w1 * p1[kb + j] + w2 * p2[kb + j];
                a[j] = (__bf16)f;
                float g = w0 * p0[kb + 16 + j] + w1 * p1[kb + 16 + j] + w2 * p2[kb + 16 + j];
                a[8 + j] = (__bf16)g;
            }
            v16bf b0 = *(const v16bf*)&Bs[((ks * 2 + 0) * 32 + lane) * 16];
            v16bf b1 = *(const v16bf*)&Bs[((ks * 2 + 1) * 32 + lane) * 16];
            acc0 = __builtin_amdgcn_wmma_f32_16x16x32_bf16(
                false, a, false, b0, (short)0, acc0, false, false);
            acc1 = __builtin_amdgcn_wmma_f32_16x16x32_bf16(
                false, a, false, b1, (short)0, acc1, false, false);
        }

        // Epilogue: C/D layout -> element r: row = rbase + r, col = cbase (+16)
        float* outp = logits + ((size_t)be * LFEAT + mt * 16 + rbase) * CCLS + cbase;
        if (mt < MTILES - 1) {
            // all 16 rows valid: branch-free stores
#pragma unroll
            for (int r = 0; r < 8; ++r) {
                outp[r * CCLS] = acc0[r] + bias0;
                if (c1ok) outp[r * CCLS + 16] = acc1[r] + bias1;
            }
        } else {
#pragma unroll
            for (int r = 0; r < 8; ++r) {
                int ll = mt * 16 + rbase + r;
                if (ll < LFEAT) {
                    outp[r * CCLS] = acc0[r] + bias0;
                    if (c1ok) outp[r * CCLS + 16] = acc1[r] + bias1;
                }
            }
        }
    }
}

// ---------------------------------------------------------------------------
// Kernel 4: mean over experts into a compact (B,196,21) buffer (L2-resident).
// ---------------------------------------------------------------------------
__global__ void mean_kernel(const float* __restrict__ logits,
                            float* __restrict__ meanL) {
    int i = blockIdx.x * 256 + threadIdx.x;
    const int per_b = LFEAT * CCLS;                 // 4116
    if (i >= BB * per_b) return;
    int lc = i % per_b;
    int b  = i / per_b;
    const float* p = logits + (size_t)b * EEXP * per_b + lc;
    meanL[i] = 0.25f * (p[0] + p[per_b] + p[2 * per_b] + p[3 * per_b]);
}

// ---------------------------------------------------------------------------
// Kernel 5: bilinear 14->224 upsample of the expert-mean.  Each thread emits
// 4 consecutive x pixels as one 16-byte store (224 % 4 == 0, aligned).
// jax.image.resize 'bilinear': src = (dst + 0.5) * (14/224) - 0.5, edge clamp.
// ---------------------------------------------------------------------------
__global__ void resize_kernel(const float* __restrict__ meanL,
                              float* __restrict__ out) {
    int t = blockIdx.x * 256 + threadIdx.x;    // one thread per 4 pixels
    // decomposition: (b, c, y, x4) with 56 groups of 4 along x
    int xg = t % 56;
    int y  = (t / 56) % IMG;
    int c  = (t / (56 * IMG)) % CCLS;
    int b  = t / (56 * IMG * CCLS);

    float sy = (y + 0.5f) * (14.0f / 224.0f) - 0.5f;
    int   y0 = (int)floorf(sy);
    float fy = sy - (float)y0;
    int y0c = min(max(y0, 0), 13), y1c = min(max(y0 + 1, 0), 13);

    const float* base = meanL + ((size_t)b * LFEAT) * CCLS + c;
    int ry0 = y0c * 14, ry1 = y1c * 14;

    float4 o;
    float res[4];
#pragma unroll
    for (int i = 0; i < 4; ++i) {
        int   x  = xg * 4 + i;
        float sx = (x + 0.5f) * (14.0f / 224.0f) - 0.5f;
        int   x0 = (int)floorf(sx);
        float fx = sx - (float)x0;
        int x0c = min(max(x0, 0), 13), x1c = min(max(x0 + 1, 0), 13);
        float t0 = (1.f - fx) * base[(ry0 + x0c) * CCLS] + fx * base[(ry0 + x1c) * CCLS];
        float t1 = (1.f - fx) * base[(ry1 + x0c) * CCLS] + fx * base[(ry1 + x1c) * CCLS];
        res[i] = (1.f - fy) * t0 + fy * t1;
    }
    o.x = res[0]; o.y = res[1]; o.z = res[2]; o.w = res[3];
    size_t off = (((size_t)b * CCLS + c) * IMG + y) * IMG + xg * 4;
    *(float4*)(out + off) = o;
}

// ---------------------------------------------------------------------------
extern "C" void kernel_launch(void* const* d_in, const int* in_sizes, int n_in,
                              void* d_out, int out_size, void* d_ws, size_t ws_size,
                              hipStream_t stream) {
    const float* features = (const float*)d_in[0];
    const float* gate_W   = (const float*)d_in[1];
    const float* gate_b   = (const float*)d_in[2];
    const float* expert_W = (const float*)d_in[3];
    const float* expert_b = (const float*)d_in[4];
    float* out = (float*)d_out;

    char* ws = (char*)d_ws;
    float* scoreBuf = (float*)(ws);              // B*NL*E = 768 floats
    int*   idxBuf   = (int*)(ws + 4096);         // B*E*K  = 192 ints
    float* wBuf     = (float*)(ws + 8192);       // B*E*K  = 192 floats
    float* logits   = (float*)(ws + 16384);      // B*E*196*21 = 263424 floats
    float* meanL    = (float*)(ws + 16384 + 263424 * sizeof(float)); // B*196*21

    gate_kernel<<<BB * NLAY, 64, 0, stream>>>(features, gate_W, gate_b, scoreBuf);
    topk_kernel<<<1, 64, 0, stream>>>(scoreBuf, idxBuf, wBuf);
    moe_gemm_kernel<<<BB * EEXP, 256, 0, stream>>>(features, expert_W, expert_b,
                                                   idxBuf, wBuf, logits);
    int meanTotal = BB * LFEAT * CCLS;                       // 65856
    mean_kernel<<<(meanTotal + 255) / 256, 256, 0, stream>>>(logits, meanL);
    int quads = BB * CCLS * IMG * IMG / 4;                   // 4,214,784
    resize_kernel<<<quads / 256, 256, 0, stream>>>(meanL, out);
}
